// GraphAttentionLayer_21414706938031
// MI455X (gfx1250) — compile-verified, matrix operand order checked by reading, not run on hardware
//
#include <hip/hip_runtime.h>
#include <hip/hip_bf16.h>

// ---------------------------------------------------------------------------
// GAT layer, MI455X (gfx1250, wave32).
//   x: (8,4096,128) f32   W: (64,128) f32   a: (1,128) f32
//   out = elu( softmax_j(lrelu(s1_i+s2_j)) @ h ),  h = x@W^T
//
// k1: h = x@W^T via v_wmma_f32_16x16x4_f32 (f32 fidelity, 0.54 GFLOP),
//     also emits hT = bf16 transpose [b][d][n] for the flash kernel.
// k2: s1/s2 scores + per-batch max(s2) (closed-form softmax max since
//     lrelu is monotone: m_i = lrelu(s1_i + max_j s2_j) -> no online rescale).
// k3: flash streaming over j: P computed in-registers (exp per lane),
//     P@H on v_wmma_f32_16x16x32_bf16 (K=32, 8x the f32-WMMA K-depth),
//     denominator via ones-B WMMA in the identical C layout; DOUBLE-BUFFERED
//     async global->LDS staging (global_load_async_to_lds_b128, ASYNCcnt
//     ping-pong: next tile in flight while current tile feeds the WMMAs).
// ---------------------------------------------------------------------------

#define NBATCH 8
#define NNODE  4096
#define IN_F   128
#define OUT_F  64
#define TJ     128          // j-tile per block
#define LPITCH 136          // LDS pitch in bf16 (272B: 16B-aligned, conflict-free)
#define CHUNKS (OUT_F * (TJ / 8))   // 1024 x 16B chunks per tile
#define CPT    (CHUNKS / 128)       // 8 async b128 issues per thread per tile

typedef __attribute__((ext_vector_type(2)))  float  v2f;
typedef __attribute__((ext_vector_type(8)))  float  v8f;
typedef __attribute__((ext_vector_type(8)))  __bf16 v8bf;
typedef __attribute__((ext_vector_type(16))) __bf16 v16bf;
typedef int vsi4 __attribute__((vector_size(16)));       // b128 payload type
typedef __attribute__((address_space(1))) vsi4* gvec_p;  // global (AS1)
typedef __attribute__((address_space(3))) vsi4* lvec_p;  // LDS (AS3)

#if defined(__has_builtin)
#if __has_builtin(__builtin_amdgcn_global_load_async_to_lds_b128)
#define HAVE_ASYNC_LDS 1
#endif
#endif

__device__ __forceinline__ float lrelu(float t) { return fmaxf(t, 0.2f * t); }

__device__ __forceinline__ v8f wmma_f32(v2f a, v2f b, v8f c) {
  return __builtin_amdgcn_wmma_f32_16x16x4_f32(false, a, false, b,
                                               (short)0, c, false, false);
}
__device__ __forceinline__ v8f wmma_bf16(v16bf a, v16bf b, v8f c) {
  return __builtin_amdgcn_wmma_f32_16x16x32_bf16(false, a, false, b,
                                                 (short)0, c, false, false);
}

// ---------------------------------------------------------------------------
// Kernel 1: h = x @ W^T (f32 WMMA) + hT bf16 transpose [b][64][4096].
// ---------------------------------------------------------------------------
__global__ __launch_bounds__(128) void k_gemm_h(const float* __restrict__ x,
                                                const float* __restrict__ W,
                                                float* __restrict__ h,
                                                __bf16* __restrict__ hT) {
  const int lane = threadIdx.x & 31;
  const int wave = threadIdx.x >> 5;
  const int hi   = lane >> 4;
  const int lm   = lane & 15;
  const long row0 = ((long)blockIdx.x * 4 + wave) * 16;

  v8f acc[4] = {{}, {}, {}, {}};
  const float* xrow = x + (row0 + lm) * IN_F + 2 * hi;

  for (int k0 = 0; k0 < IN_F; k0 += 4) {
    v2f af = *(const v2f*)(xrow + k0);
#pragma unroll
    for (int t = 0; t < 4; ++t) {
      v2f bf = *(const v2f*)(W + (t * 16 + lm) * IN_F + k0 + 2 * hi);
      acc[t] = wmma_f32(af, bf, acc[t]);
    }
  }
  const long bb = row0 >> 12;            // batch
  const int  n0 = (int)(row0 & 4095);    // node within batch
#pragma unroll
  for (int t = 0; t < 4; ++t)
#pragma unroll
    for (int r = 0; r < 8; ++r) {
      float v = acc[t][r];
      h[(row0 + r + 8 * hi) * OUT_F + t * 16 + lm] = v;
      hT[(bb * OUT_F + t * 16 + lm) * (long)NNODE + n0 + r + 8 * hi] = (__bf16)v;
    }
}

// ---------------------------------------------------------------------------
// Kernel 2: s1 = h.a1, s2 = h.a2, per-batch max(s2). One block per batch.
// ---------------------------------------------------------------------------
__global__ __launch_bounds__(256) void k_scores(const float* __restrict__ h,
                                                const float* __restrict__ a,
                                                float* __restrict__ s1,
                                                float* __restrict__ s2,
                                                float* __restrict__ s2max) {
  __shared__ float red[256];
  const int b = blockIdx.x;
  float lmax = -3.0e38f;
  for (int it = 0; it < NNODE / 256; ++it) {
    const long row = (long)b * NNODE + it * 256 + threadIdx.x;
    const float* hr = h + row * OUT_F;
    float d1 = 0.f, d2 = 0.f;
#pragma unroll 8
    for (int d = 0; d < OUT_F; ++d) {
      float v = hr[d];
      d1 = fmaf(v, a[d], d1);
      d2 = fmaf(v, a[OUT_F + d], d2);
    }
    s1[row] = d1;
    s2[row] = d2;
    lmax = fmaxf(lmax, d2);
  }
  red[threadIdx.x] = lmax;
  __syncthreads();
  for (int s = 128; s > 0; s >>= 1) {
    if (threadIdx.x < s)
      red[threadIdx.x] = fmaxf(red[threadIdx.x], red[threadIdx.x + s]);
    __syncthreads();
  }
  if (threadIdx.x == 0) s2max[b] = red[0];
}

// ---------------------------------------------------------------------------
// Kernel 3: flash attention on bf16 WMMA with double-buffered async staging.
// Block = 64 i-rows x 1 batch, 4 waves of 16 rows. hT tile [64 d][128 j] in
// LDS (bf16, pitch 136, ping-pong). Per K-macro-step (32 j): A-frag = 16
// in-lane exps -> v16bf; B-frag = two ds_load_b128 per d-tile; 5 WMMAs
// (4 d-tiles + denominator in the identical C layout).
// ---------------------------------------------------------------------------
__global__ __launch_bounds__(128) void k_attn(const __bf16* __restrict__ hT,
                                              const float* __restrict__ s1g,
                                              const float* __restrict__ s2g,
                                              const float* __restrict__ s2maxg,
                                              float* __restrict__ out) {
  __shared__ __align__(16) __bf16 hjT[2][OUT_F * LPITCH];  // ~35 KB ping-pong
  __shared__ float s2t[TJ];

  const int b    = blockIdx.y;
  const int lane = threadIdx.x & 31;
  const int wave = threadIdx.x >> 5;
  const int hi   = lane >> 4;
  const int lm   = lane & 15;
  const long bbase = (long)b * NNODE;
  const int i0   = blockIdx.x * 64 + wave * 16;

  const float s1v = s1g[bbase + i0 + lm];        // A-frag row = lane%16
  const float m   = lrelu(s1v + s2maxg[b]);

  v8f acc[4] = {{}, {}, {}, {}};
  v8f accd   = {};
  v16bf ones;
#pragma unroll
  for (int e = 0; e < 16; ++e) ones[e] = (__bf16)1.0f;

  const int kb0 = hi ? 8 : 0;    // this lane's K-blocks within a 32-step
  const __bf16* hTb = hT + (long)b * OUT_F * NNODE;

  // stage tile [jtile, jtile+TJ) of hT into LDS buffer `buf` (async if avail)
  auto stage = [&](int jtile, int buf) {
#pragma unroll
    for (int q = 0; q < CPT; ++q) {
      const int c = threadIdx.x + q * 128;       // 16B chunk id
      const int row = c >> 4, col8 = c & 15;
      const __bf16* g = hTb + (long)row * NNODE + jtile + col8 * 8;
      __bf16* l = &hjT[buf][row * LPITCH + col8 * 8];
#ifdef HAVE_ASYNC_LDS
      __builtin_amdgcn_global_load_async_to_lds_b128((gvec_p)g, (lvec_p)l, 0, 0);
#else
      *(v8bf*)l = *(const v8bf*)g;
#endif
    }
  };

  stage(0, 0);                                   // preload first tile

  for (int jt = 0, it = 0; jt < NNODE; jt += TJ, ++it) {
    const int buf = it & 1;
    const bool hasNext = (jt + TJ) < NNODE;
    if (hasNext) stage(jt + TJ, buf ^ 1);        // next tile in flight
    s2t[threadIdx.x] = s2g[bbase + jt + threadIdx.x];
#ifdef HAVE_ASYNC_LDS
    // current tile's loads complete in order; leave only next tile's (<=CPT)
    if (hasNext) asm volatile("s_wait_asynccnt %0" ::"i"(CPT) : "memory");
    else         asm volatile("s_wait_asynccnt 0" ::: "memory");
#endif
    __syncthreads();

    const __bf16* tile = hjT[buf];
#pragma unroll
    for (int j0 = 0; j0 < TJ; j0 += 32) {
      // A-fragment: p = exp(lrelu(s1+s2) - m), 16 values of this lane's row
      v16bf af;
#pragma unroll
      for (int e = 0; e < 8; ++e) {
        af[e]     = (__bf16)__expf(lrelu(s1v + s2t[j0 + kb0 + e])      - m);
        af[e + 8] = (__bf16)__expf(lrelu(s1v + s2t[j0 + kb0 + 16 + e]) - m);
      }
#pragma unroll
      for (int t = 0; t < 4; ++t) {
        const __bf16* lrow = tile + (t * 16 + lm) * LPITCH + j0 + kb0;
        v8bf blo = *(const v8bf*)(lrow);
        v8bf bhi = *(const v8bf*)(lrow + 16);
        v16bf bf = __builtin_shufflevector(blo, bhi, 0, 1, 2, 3, 4, 5, 6, 7,
                                           8, 9, 10, 11, 12, 13, 14, 15);
        acc[t] = wmma_bf16(af, bf, acc[t]);
      }
      accd = wmma_bf16(af, ones, accd);          // row-sums, same C layout
    }
    __syncthreads();                             // readers done before reuse
  }

#pragma unroll
  for (int t = 0; t < 4; ++t)
#pragma unroll
    for (int r = 0; r < 8; ++r) {
      float v = acc[t][r] / accd[r];
      v = v > 0.f ? v : (__expf(v) - 1.f);       // elu
      out[(bbase + i0 + r + 8 * hi) * OUT_F + t * 16 + lm] = v;
    }
}

// ---------------------------------------------------------------------------
extern "C" void kernel_launch(void* const* d_in, const int* in_sizes, int n_in,
                              void* d_out, int out_size, void* d_ws,
                              size_t ws_size, hipStream_t stream) {
  const float* x = (const float*)d_in[0];  // 8*4096*128
  const float* W = (const float*)d_in[1];  // 64*128
  const float* a = (const float*)d_in[2];  // 128
  float* out = (float*)d_out;              // 8*4096*64

  const long NR = (long)NBATCH * NNODE;    // 32768 rows
  float* h     = (float*)d_ws;             // NR*64 f32 (8.4 MB)
  float* s1    = h + NR * OUT_F;
  float* s2    = s1 + NR;
  float* s2max = s2 + NR;
  __bf16* hT   = (__bf16*)(s2max + 16);    // NR*64 bf16 (4.2 MB), 16B aligned

  k_gemm_h<<<dim3(NR / 64), dim3(128), 0, stream>>>(x, W, h, hT);
  k_scores<<<dim3(NBATCH), dim3(256), 0, stream>>>(h, a, s1, s2, s2max);
  k_attn<<<dim3(NNODE / 64, NBATCH), dim3(128), 0, stream>>>(hT, s1, s2,
                                                             s2max, out);
}
